// MultiHeadAttention_32607391711957
// MI455X (gfx1250) — compile-verified, matrix operand order checked by reading, not run on hardware
//
#include <hip/hip_runtime.h>
#include <hip/hip_bf16.h>

// ---------------------------------------------------------------------------
// MHA (B=1, T=1500, N=1280, H=20, D=64) via CDNA5 WMMA f16 (f32 accumulate).
// v2: double-buffered (software-pipelined) fragment loads, 32x64 GEMM strips.
// ---------------------------------------------------------------------------

#define T_LEN   1500
#define T_PAD   1504            // 94 tiles of 16
#define NSTATE  1280
#define NHEAD   20
#define DHEAD   64
#define QKSCALE 0.125f          // 64^-0.5

typedef __attribute__((ext_vector_type(16))) _Float16 v16h;
typedef __attribute__((ext_vector_type(8)))  _Float16 v8h;
typedef __attribute__((ext_vector_type(8)))  float    v8f;

// ---------------------------------------------------------------------------
// Fragment loader: 16-bit A/B layout (wave32).
// lane L holds row (A) / column (B) = L&15.
// element e -> K index = e + 8*(e>=8) + 8*(L>=16)
// => two contiguous 8-half (16B) chunks at k0+8*hi and k0+16+8*hi.
// ---------------------------------------------------------------------------
__device__ __forceinline__ v16h load_frag(const _Float16* __restrict__ base,
                                          int ld, int row, int k0, int hi) {
  const _Float16* p = base + (long)row * ld + k0 + 8 * hi;
  v8h lo = *(const v8h*)(p);
  v8h hv = *(const v8h*)(p + 16);
  v16h a;
#pragma unroll
  for (int i = 0; i < 8; ++i) { a[i] = lo[i]; a[i + 8] = hv[i]; }
  return a;
}

__device__ __forceinline__ v8f wmma_f16(v16h a, v16h b, v8f c) {
  return __builtin_amdgcn_wmma_f32_16x16x32_f16(false, a, false, b,
                                                (short)0, c, false, false);
}

__device__ __forceinline__ v8f vzero8() {
  return (v8f){0.f, 0.f, 0.f, 0.f, 0.f, 0.f, 0.f, 0.f};
}

// ---------------------------------------------------------------------------
// f32 -> f16 convert with optional zero row padding.
// ---------------------------------------------------------------------------
__global__ void convert_f32_f16_pad(const float* __restrict__ src,
                                    _Float16* __restrict__ dst,
                                    int rows, int cols, int rowsPad) {
  int i = blockIdx.x * 256 + threadIdx.x;
  int total = rowsPad * cols;
  if (i >= total) return;
  int r = i / cols;
  dst[i] = (r < rows) ? (_Float16)src[i] : (_Float16)0.0f;
}

// ---------------------------------------------------------------------------
// GEMM: C[M,N] = scale * (A[M,K] @ W[N,K]^T) + bias
// One wave computes a 32x64 strip: 2 A-frags x 4 B-frags -> 8 WMMAs per
// 32-wide K step. K loop is unrolled by 2 with ping-pong fragment buffers so
// loads for step k+1 overlap WMMAs of step k. Requires K % 64 == 0.
// ---------------------------------------------------------------------------
__global__ __launch_bounds__(256)
void gemm16_wmma(const _Float16* __restrict__ A, int lda,
                 const _Float16* __restrict__ W, int ldw,
                 const float* __restrict__ bias, float scale,
                 int Mpad, int N, int K,
                 _Float16* __restrict__ out16, int ldc16, int transposed, int ldt,
                 float* __restrict__ out32, int ldc32, int Mstore) {
  int wave  = (blockIdx.x * blockDim.x + threadIdx.x) >> 5;
  int lane  = threadIdx.x & 31;
  int nGrp  = N >> 6;                        // groups of 64 columns
  int tileM = wave / nGrp;                   // 32-row tiles
  int grpN  = wave % nGrp;
  if (tileM * 32 >= Mpad) return;            // wave-uniform

  int hi = lane >> 4;
  int lr = lane & 15;
  int ar0 = tileM * 32 + lr;
  int ar1 = ar0 + 16;
  int wr[4];
#pragma unroll
  for (int t = 0; t < 4; ++t) wr[t] = grpN * 64 + t * 16 + lr;

  v8f acc[2][4];
#pragma unroll
  for (int mi = 0; mi < 2; ++mi)
#pragma unroll
    for (int t = 0; t < 4; ++t) acc[mi][t] = vzero8();

  v16h a0[2], b0[4], a1[2], b1[4];

  // preload buffer 0 @ k = 0
  a0[0] = load_frag(A, lda, ar0, 0, hi);
  a0[1] = load_frag(A, lda, ar1, 0, hi);
#pragma unroll
  for (int t = 0; t < 4; ++t) b0[t] = load_frag(W, ldw, wr[t], 0, hi);

  int k0;
  for (k0 = 0; k0 + 64 < K; k0 += 64) {
    // prefetch buffer 1 @ k0+32
    a1[0] = load_frag(A, lda, ar0, k0 + 32, hi);
    a1[1] = load_frag(A, lda, ar1, k0 + 32, hi);
#pragma unroll
    for (int t = 0; t < 4; ++t) b1[t] = load_frag(W, ldw, wr[t], k0 + 32, hi);
    // consume buffer 0 @ k0
#pragma unroll
    for (int mi = 0; mi < 2; ++mi)
#pragma unroll
      for (int t = 0; t < 4; ++t)
        acc[mi][t] = wmma_f16(a0[mi], b0[t], acc[mi][t]);
    // prefetch buffer 0 @ k0+64
    a0[0] = load_frag(A, lda, ar0, k0 + 64, hi);
    a0[1] = load_frag(A, lda, ar1, k0 + 64, hi);
#pragma unroll
    for (int t = 0; t < 4; ++t) b0[t] = load_frag(W, ldw, wr[t], k0 + 64, hi);
    // consume buffer 1 @ k0+32
#pragma unroll
    for (int mi = 0; mi < 2; ++mi)
#pragma unroll
      for (int t = 0; t < 4; ++t)
        acc[mi][t] = wmma_f16(a1[mi], b1[t], acc[mi][t]);
  }
  // tail: k0 == K-64; buffer 0 holds K-64, load buffer 1 @ K-32
  a1[0] = load_frag(A, lda, ar0, K - 32, hi);
  a1[1] = load_frag(A, lda, ar1, K - 32, hi);
#pragma unroll
  for (int t = 0; t < 4; ++t) b1[t] = load_frag(W, ldw, wr[t], K - 32, hi);
#pragma unroll
  for (int mi = 0; mi < 2; ++mi)
#pragma unroll
    for (int t = 0; t < 4; ++t)
      acc[mi][t] = wmma_f16(a0[mi], b0[t], acc[mi][t]);
#pragma unroll
  for (int mi = 0; mi < 2; ++mi)
#pragma unroll
    for (int t = 0; t < 4; ++t)
      acc[mi][t] = wmma_f16(a1[mi], b1[t], acc[mi][t]);

  // ---- epilogue ----
#pragma unroll
  for (int mi = 0; mi < 2; ++mi) {
#pragma unroll
    for (int t = 0; t < 4; ++t) {
      int col = grpN * 64 + t * 16 + lr;
      float b = bias ? bias[col] : 0.0f;
      float vals[8];
#pragma unroll
      for (int r = 0; r < 8; ++r) vals[r] = acc[mi][t][r] * scale + b;

      if (out16) {
        if (transposed) {
          v8h pk;
#pragma unroll
          for (int r = 0; r < 8; ++r) pk[r] = (_Float16)vals[r];
          *(v8h*)(out16 + (long)col * ldt + tileM * 32 + mi * 16 + 8 * hi) = pk;
        } else {
#pragma unroll
          for (int r = 0; r < 8; ++r) {
            int m = tileM * 32 + mi * 16 + r + 8 * hi;
            out16[(long)m * ldc16 + col] = (_Float16)vals[r];
          }
        }
      }
      if (out32) {
#pragma unroll
        for (int r = 0; r < 8; ++r) {
          int m = tileM * 32 + mi * 16 + r + 8 * hi;
          if (m < Mstore) out32[(long)m * ldc32 + col] = vals[r];
        }
      }
    }
  }
}

// ---------------------------------------------------------------------------
// Attention: one wave per (16-query tile, head).
// scores (f16) held in LDS [16][T_PAD]; softmax in f32; PV via WMMA with
// A-fragments streamed straight out of LDS and B-fragments from V^T.
// K-tile and V-tile global fragment loads are double-buffered.
// ---------------------------------------------------------------------------
__global__ __launch_bounds__(32)
void attn_wmma(const _Float16* __restrict__ q16,
               const _Float16* __restrict__ k16,
               const _Float16* __restrict__ v16t,   // [NSTATE, T_PAD]
               _Float16* __restrict__ wv16) {       // [T_PAD, NSTATE]
  __shared__ __align__(16) _Float16 sc[16][T_PAD];
  __shared__ float inv_s[16];

  int lane  = threadIdx.x;
  int hi    = lane >> 4;
  int lr    = lane & 15;
  int qtile = blockIdx.x;            // 0..93
  int head  = blockIdx.y;            // 0..19
  int hbase = head * DHEAD;

  // Q fragments for this 16-row tile (K already carries the 1/8 scale).
  int qrow = qtile * 16 + lr;
  v16h qa0 = load_frag(q16, NSTATE, qrow, hbase,      hi);
  v16h qa1 = load_frag(q16, NSTATE, qrow, hbase + 32, hi);

  // ---- scores = Q @ K^T -> LDS (f16), double-buffered over key tiles ----
  v16h kA0, kA1, kB0, kB1;
  kA0 = load_frag(k16, NSTATE, lr, hbase,      hi);
  kA1 = load_frag(k16, NSTATE, lr, hbase + 32, hi);
  const int NKT = T_PAD / 16;        // 94 (even)
  for (int kt = 0; kt + 2 < NKT; kt += 2) {
    int krowB = (kt + 1) * 16 + lr;
    kB0 = load_frag(k16, NSTATE, krowB, hbase,      hi);
    kB1 = load_frag(k16, NSTATE, krowB, hbase + 32, hi);
    {
      v8f s = vzero8();
      s = wmma_f16(qa0, kA0, s);
      s = wmma_f16(qa1, kA1, s);
#pragma unroll
      for (int r = 0; r < 8; ++r)
        sc[r + 8 * hi][kt * 16 + lr] = (_Float16)s[r];
    }
    int krowA = (kt + 2) * 16 + lr;
    kA0 = load_frag(k16, NSTATE, krowA, hbase,      hi);
    kA1 = load_frag(k16, NSTATE, krowA, hbase + 32, hi);
    {
      v8f s = vzero8();
      s = wmma_f16(qa0, kB0, s);
      s = wmma_f16(qa1, kB1, s);
#pragma unroll
      for (int r = 0; r < 8; ++r)
        sc[r + 8 * hi][(kt + 1) * 16 + lr] = (_Float16)s[r];
    }
  }
  // tail: tiles NKT-2 (in buffer A) and NKT-1
  {
    int krowB = (NKT - 1) * 16 + lr;
    kB0 = load_frag(k16, NSTATE, krowB, hbase,      hi);
    kB1 = load_frag(k16, NSTATE, krowB, hbase + 32, hi);
    v8f s = vzero8();
    s = wmma_f16(qa0, kA0, s);
    s = wmma_f16(qa1, kA1, s);
#pragma unroll
    for (int r = 0; r < 8; ++r)
      sc[r + 8 * hi][(NKT - 2) * 16 + lr] = (_Float16)s[r];
    v8f s2 = vzero8();
    s2 = wmma_f16(qa0, kB0, s2);
    s2 = wmma_f16(qa1, kB1, s2);
#pragma unroll
    for (int r = 0; r < 8; ++r)
      sc[r + 8 * hi][(NKT - 1) * 16 + lr] = (_Float16)s2[r];
  }
  __syncthreads();

  // ---- softmax over keys [0, T_LEN): two lanes per row ----
  float mx = -1e30f;
  for (int k = hi; k < T_LEN; k += 2) mx = fmaxf(mx, (float)sc[lr][k]);
  mx = fmaxf(mx, __shfl_xor(mx, 16, 32));

  float sum = 0.0f;
  for (int k = hi; k < T_LEN; k += 2) {
    float p = __expf((float)sc[lr][k] - mx);
    sum += p;
    sc[lr][k] = (_Float16)p;
  }
  // zero padded key columns so they contribute nothing to PV
  for (int k = T_LEN + hi; k < T_PAD; k += 2) sc[lr][k] = (_Float16)0.0f;
  sum += __shfl_xor(sum, 16, 32);
  if (hi == 0) inv_s[lr] = 1.0f / sum;
  __syncthreads();

  // ---- O = P @ V (P from LDS, V^T contiguous), V frags double-buffered ----
  v8f o[4];
#pragma unroll
  for (int t = 0; t < 4; ++t) o[t] = vzero8();

  v16h vA[4], vB[4];
#pragma unroll
  for (int t = 0; t < 4; ++t)
    vA[t] = load_frag(v16t, T_PAD, hbase + t * 16 + lr, 0, hi);

  int k0;
  for (k0 = 0; k0 + 64 < T_PAD; k0 += 64) {   // T_PAD = 47 * 32: odd step count
#pragma unroll
    for (int t = 0; t < 4; ++t)
      vB[t] = load_frag(v16t, T_PAD, hbase + t * 16 + lr, k0 + 32, hi);
    {
      v16h pa = load_frag(&sc[0][0], T_PAD, lr, k0, hi);
#pragma unroll
      for (int t = 0; t < 4; ++t) o[t] = wmma_f16(pa, vA[t], o[t]);
    }
#pragma unroll
    for (int t = 0; t < 4; ++t)
      vA[t] = load_frag(v16t, T_PAD, hbase + t * 16 + lr, k0 + 64, hi);
    {
      v16h pa = load_frag(&sc[0][0], T_PAD, lr, k0 + 32, hi);
#pragma unroll
      for (int t = 0; t < 4; ++t) o[t] = wmma_f16(pa, vB[t], o[t]);
    }
  }
  // tail: final 32-wide step @ k0 == T_PAD-32 (buffer A already loaded)
  {
    v16h pa = load_frag(&sc[0][0], T_PAD, lr, T_PAD - 32, hi);
#pragma unroll
    for (int t = 0; t < 4; ++t) o[t] = wmma_f16(pa, vA[t], o[t]);
  }

  // scale by 1/sum and store f16 into wv16 [T_PAD, NSTATE]
#pragma unroll
  for (int t = 0; t < 4; ++t) {
#pragma unroll
    for (int r = 0; r < 8; ++r) {
      int m = qtile * 16 + r + 8 * hi;
      float val = o[t][r] * inv_s[r + 8 * hi];
      wv16[(long)m * NSTATE + hbase + t * 16 + lr] = (_Float16)val;
    }
  }
}

// ---------------------------------------------------------------------------
// Launch
// ---------------------------------------------------------------------------
extern "C" void kernel_launch(void* const* d_in, const int* in_sizes, int n_in,
                              void* d_out, int out_size, void* d_ws, size_t ws_size,
                              hipStream_t stream) {
  const float* x  = (const float*)d_in[0];
  const float* Wq = (const float*)d_in[1];
  const float* bq = (const float*)d_in[2];
  const float* Wk = (const float*)d_in[3];
  const float* Wv = (const float*)d_in[4];
  const float* bv = (const float*)d_in[5];
  const float* Wo = (const float*)d_in[6];
  const float* bo = (const float*)d_in[7];
  float* out = (float*)d_out;

  const size_t SZ_TN = (size_t)T_PAD * NSTATE * sizeof(_Float16);   // 3.85 MB
  const size_t SZ_WW = (size_t)NSTATE * NSTATE * sizeof(_Float16);  // 3.28 MB
  char* ws = (char*)d_ws;
  _Float16* x16  = (_Float16*)(ws);                 ws += SZ_TN;
  _Float16* q16  = (_Float16*)(ws);                 ws += SZ_TN;
  _Float16* k16  = (_Float16*)(ws);                 ws += SZ_TN;
  _Float16* v16t = (_Float16*)(ws);                 ws += SZ_TN;   // [NSTATE, T_PAD]
  _Float16* wv16 = (_Float16*)(ws);                 ws += SZ_TN;
  _Float16* wq16 = (_Float16*)(ws);                 ws += SZ_WW;
  _Float16* wk16 = (_Float16*)(ws);                 ws += SZ_WW;
  _Float16* wv16w= (_Float16*)(ws);                 ws += SZ_WW;
  _Float16* wo16 = (_Float16*)(ws);                 ws += SZ_WW;

  // --- converts ---
  {
    int totalX = T_PAD * NSTATE;
    convert_f32_f16_pad<<<(totalX + 255) / 256, 256, 0, stream>>>(x, x16, T_LEN, NSTATE, T_PAD);
    int totalW = NSTATE * NSTATE;
    int bw = (totalW + 255) / 256;
    convert_f32_f16_pad<<<bw, 256, 0, stream>>>(Wq, wq16,  NSTATE, NSTATE, NSTATE);
    convert_f32_f16_pad<<<bw, 256, 0, stream>>>(Wk, wk16,  NSTATE, NSTATE, NSTATE);
    convert_f32_f16_pad<<<bw, 256, 0, stream>>>(Wv, wv16w, NSTATE, NSTATE, NSTATE);
    convert_f32_f16_pad<<<bw, 256, 0, stream>>>(Wo, wo16,  NSTATE, NSTATE, NSTATE);
  }

  // --- projection GEMMs: 47 (32-row) M-tiles x 20 column-groups = 940 waves ---
  const int tilesM = T_PAD / 32;             // 47
  const int nGrp   = NSTATE / 64;            // 20
  const int waves  = tilesM * nGrp;          // 940
  const int blocks = (waves * 32 + 255) / 256;

  // Q = x @ Wq^T + bq
  gemm16_wmma<<<blocks, 256, 0, stream>>>(x16, NSTATE, wq16, NSTATE, bq, 1.0f,
                                          T_PAD, NSTATE, NSTATE,
                                          q16, NSTATE, 0, 0, nullptr, 0, 0);
  // K = (x @ Wk^T) * QKSCALE   (scale folded here)
  gemm16_wmma<<<blocks, 256, 0, stream>>>(x16, NSTATE, wk16, NSTATE, nullptr, QKSCALE,
                                          T_PAD, NSTATE, NSTATE,
                                          k16, NSTATE, 0, 0, nullptr, 0, 0);
  // V = x @ Wv^T + bv, stored transposed [NSTATE, T_PAD]
  gemm16_wmma<<<blocks, 256, 0, stream>>>(x16, NSTATE, wv16w, NSTATE, bv, 1.0f,
                                          T_PAD, NSTATE, NSTATE,
                                          v16t, 0, 1, T_PAD, nullptr, 0, 0);

  // --- attention ---
  attn_wmma<<<dim3(T_PAD / 16, NHEAD), 32, 0, stream>>>(q16, k16, v16t, wv16);

  // --- output projection: out = wv @ Wo^T + bo (f32, guarded to 1500 rows) ---
  gemm16_wmma<<<blocks, 256, 0, stream>>>(wv16, NSTATE, wo16, NSTATE, bo, 1.0f,
                                          T_PAD, NSTATE, NSTATE,
                                          nullptr, 0, 0, 0, out, NSTATE, T_LEN);

  (void)in_sizes; (void)n_in; (void)out_size; (void)ws_size;
}
